// IntentDetectionOnlyAttentionModel_26173530702468
// MI455X (gfx1250) — matrix operand order repaired and checked by reading, not compiled, not run on hardware
//
#include <hip/hip_runtime.h>
#include <hip/hip_bf16.h>

// Problem constants (from reference): B=256, S=128, H=512, OUT=64, L=3
//   4H = 2048 features (GEMM K and N), 2H = 1024, ROWS = B*S = 32768.

typedef __attribute__((ext_vector_type(16))) __bf16 bf16x16;
typedef __attribute__((ext_vector_type(8)))  float  v8f;
typedef __attribute__((ext_vector_type(4)))  unsigned int v4u;
typedef __attribute__((ext_vector_type(8)))  int v8i;
typedef __attribute__((ext_vector_type(4)))  int v4i;

// Native f32 -> bf16 (lets the backend emit v_cvt_pk_bf16_f32 instead of
// a scalarized round-to-nearest-even bit sequence).
__device__ __forceinline__ unsigned short f2bf(float x) {
    union { __bf16 b; unsigned short u; } c;
    c.b = (__bf16)x;
    return c.u;
}
__device__ __forceinline__ unsigned pack_bf2(float lo, float hi) {
    union { __bf16 b[2]; unsigned u; } c;
    c.b[0] = (__bf16)lo;
    c.b[1] = (__bf16)hi;
    return c.u;
}

// ---------------------------------------------------------------------------
// 0) init: zero BN1 accumulators + energy
__global__ __launch_bounds__(256) void init_kernel(float* sums, float* sumsq, float* energy) {
    int i = blockIdx.x * 256 + threadIdx.x;           // 128 blocks -> 32768
    if (i < 2048) { sums[i] = 0.f; sumsq[i] = 0.f; }
    if (i < 32768) energy[i] = 0.f;
}

// 1) sent_emb[b][j] from last LSTM layer hidden state (planes 4,5 of (6,B,H))
__global__ __launch_bounds__(256) void sent_kernel(const float* __restrict__ hs, float* __restrict__ sent) {
    int idx = blockIdx.x * 256 + threadIdx.x;         // 1024 blocks -> 256*1024
    int b = idx >> 10, j = idx & 1023;
    int plane = 4 + (j >> 9), h = j & 511;
    sent[idx] = hs[((size_t)plane * 256 + b) * 512 + h];
}

// 2a) masked BN1 stats for word features (f in [0,1024))
__global__ __launch_bounds__(256) void bn1_words_stats(const float* __restrict__ lstm,
                                                       const int* __restrict__ lens,
                                                       float* sums, float* sumsq) {
    int blk = blockIdx.x;                              // 256 blocks = one batch each
    int t = threadIdx.x;
    int len = lens[blk];
    float s[4] = {0, 0, 0, 0}, q[4] = {0, 0, 0, 0};
    for (int r = 0; r < len; ++r) {
        const float* p = lstm + ((size_t)blk * 128 + r) * 1024;
#pragma unroll
        for (int i = 0; i < 4; ++i) { float v = p[t + 256 * i]; s[i] += v; q[i] += v * v; }
    }
#pragma unroll
    for (int i = 0; i < 4; ++i) {
        atomicAdd(&sums[t + 256 * i], s[i]);
        atomicAdd(&sumsq[t + 256 * i], q[i]);
    }
}

// 2b) masked BN1 stats for sent features (f in [1024,2048)): weight by seq length
__global__ __launch_bounds__(256) void bn1_sent_stats(const float* __restrict__ sent,
                                                      const int* __restrict__ lens,
                                                      float* sums, float* sumsq, float* cntf) {
    int f = blockIdx.x * 256 + threadIdx.x;            // 4 blocks -> 1024
    float s = 0.f, q = 0.f;
    for (int b = 0; b < 256; ++b) {
        float l = (float)lens[b];
        float v = sent[(size_t)b * 1024 + f];
        s += v * l; q += v * v * l;
    }
    sums[1024 + f] = s; sumsq[1024 + f] = q;
    if (blockIdx.x == 0 && threadIdx.x == 0) {
        int c = 0;
        for (int b = 0; b < 256; ++b) c += lens[b];
        cntf[0] = (float)c;
    }
}

// 3) BN1 -> per-feature affine: scale1/shift1
__global__ __launch_bounds__(256) void bn1_finalize(const float* sums, const float* sumsq, const float* cntf,
                                                    const float* __restrict__ g, const float* __restrict__ b,
                                                    float* scale1, float* shift1) {
    int f = blockIdx.x * 256 + threadIdx.x;            // 8 blocks -> 2048
    float cnt = cntf[0];
    float mean = sums[f] / cnt;
    float var = sumsq[f] / cnt - mean * mean;
    float sc = g[f] * rsqrtf(var + 1e-5f);
    scale1[f] = sc;
    shift1[f] = b[f] - mean * sc;
}

// 4a) W1' = diag(scale1) * W1 in bf16, pre-swizzled to WMMA B-fragment order:
//     elem (f,g): kc=f>>5, kl=f&31, lane=(g&15)+16*(kl>>4), j=kl&15, ntg=g>>4
//     addr = (kc*128+ntg)*512 + lane*16 + j
__global__ __launch_bounds__(256) void fold_w1_kernel(const float* __restrict__ W1,
                                                      const float* __restrict__ scale1,
                                                      unsigned short* __restrict__ W1s) {
    int idx = blockIdx.x * 256 + threadIdx.x;          // 1024 blocks: 2048 f * 128 g-groups
    int f = idx >> 7;
    int g0 = (idx & 127) << 4;
    float s = scale1[f];
    int kc = f >> 5, kl = f & 31, khalf = kl >> 4, j = kl & 15;
    int ntg = g0 >> 4;
    size_t dbase = ((size_t)(kc * 128 + ntg)) * 512 + (size_t)khalf * 256 + j;
    const float* src = W1 + (size_t)f * 2048 + g0;
#pragma unroll
    for (int n = 0; n < 16; ++n) W1s[dbase + (size_t)n * 16] = f2bf(s * src[n]);
}

// 4b) b1' = shift1 @ W1 + b1
__global__ __launch_bounds__(256) void fold_b1_kernel(const float* __restrict__ W1,
                                                      const float* __restrict__ shift1,
                                                      const float* __restrict__ b1, float* b1p) {
    int g = blockIdx.x * 256 + threadIdx.x;            // 8 blocks -> 2048
    float acc = b1[g];
    for (int f = 0; f < 2048; ++f) acc += shift1[f] * W1[(size_t)f * 2048 + g];
    b1p[g] = acc;
}

// ---------------------------------------------------------------------------
// 5) Fused mega-GEMM: energy += sum_n tanh( (x @ W1')[row,n] + b1'[n] ) * W2[n]
//    Block = 128 rows (one batch) x 128 cols; 8 waves (4 M-split x 2 N-split),
//    each wave owns 2x4 tiles of 16x16; K = 2048 streamed in 32-chunks,
//    LDS double buffered; B staged by TDM (tensor_load_to_lds), A by ds stores.
__global__ __launch_bounds__(256) void attn_gemm_kernel(const float* __restrict__ lstm,
                                                        const float* __restrict__ sent,
                                                        const unsigned short* __restrict__ W1s,
                                                        const float* __restrict__ W2,
                                                        const float* __restrict__ b1p,
                                                        float* __restrict__ energy) {
    __shared__ __align__(16) unsigned short a_sh[2][128 * 32];   // 8KB x2
    __shared__ __align__(16) unsigned short b_sh[2][8 * 32 * 16];// 8KB x2 (fragment order)
    __shared__ float e_part[128];

    const int tid = threadIdx.x;
    const int wave = tid >> 5, lane = tid & 31;
    const int wave_m = wave & 3, wave_n = wave >> 2;
    const int by = blockIdx.y;            // M block index == batch index (S==128)
    const int nb = blockIdx.x;            // N block index (0..15)
    const int row0 = by * 128;

    if (tid < 128) e_part[tid] = 0.f;

    const unsigned lds_b[2] = { (unsigned)(size_t)&b_sh[0][0],
                                (unsigned)(size_t)&b_sh[1][0] };

    auto tdm_issue = [&](int kc, int buf) {
        // 1-D copy of 8KB (this kc, this 8-ntile N block) into b_sh[buf]
        unsigned long long ga =
            (unsigned long long)(size_t)(W1s + ((size_t)kc * 128 + (size_t)nb * 8) * 512);
        v4u g0;
        g0[0] = 1u;                                         // count=1, user descriptor
        g0[1] = lds_b[buf];                                 // lds_addr
        g0[2] = (unsigned)(ga & 0xFFFFFFFFull);             // global_addr[31:0]
        g0[3] = ((unsigned)(ga >> 32) & 0x01FFFFFFu) | (2u << 30); // addr[56:32] | type=2
        v8i g1;
        g1[0] = (int)(3u << 16);                            // data_size = 8B
        g1[1] = (int)(1024u << 16);                         // tensor_dim0 = 1024 units
        g1[2] = (int)(1u << 16);                            // tensor_dim1 = 1
        g1[3] = (int)(1024u << 16);                         // tile_dim0 = 1024 units (8KB)
        g1[4] = 0;                                          // tile_dim1/2 unused
        g1[5] = 1024;                                       // tensor_dim0_stride
        g1[6] = 0; g1[7] = 0;
        v4i gz4 = {0, 0, 0, 0};
        v8i gz8 = {0, 0, 0, 0, 0, 0, 0, 0};
        // 6-arg toolchain form: (g0, g1, g2, g3, g4, cpol)
        __builtin_amdgcn_tensor_load_to_lds(g0, g1, gz4, gz4, gz8, 0);
    };

    auto loadA = [&](int kc, float4* r) {
        int m = tid >> 1, klb = (tid & 1) * 16;
        int f0 = kc * 32 + klb;
        const float* src = (f0 < 1024)
            ? (lstm + ((size_t)(row0 + m)) * 1024 + f0)        // words part (row-major)
            : (sent + (size_t)by * 1024 + (f0 - 1024));        // sent part (row-constant)
        const float4* p = reinterpret_cast<const float4*>(src);
        r[0] = p[0]; r[1] = p[1]; r[2] = p[2]; r[3] = p[3];
    };

    auto storeA = [&](const float4* r, int buf) {
        int m = tid >> 1, klb = (tid & 1) * 16;
        const float* fp = reinterpret_cast<const float*>(r);
        unsigned u[8];
#pragma unroll
        for (int i = 0; i < 8; ++i)
            u[i] = pack_bf2(fp[2 * i], fp[2 * i + 1]);     // v_cvt_pk_bf16_f32 path
        *reinterpret_cast<uint4*>(&a_sh[buf][m * 32 + klb])     = make_uint4(u[0], u[1], u[2], u[3]);
        *reinterpret_cast<uint4*>(&a_sh[buf][m * 32 + klb + 8]) = make_uint4(u[4], u[5], u[6], u[7]);
    };

    union FragBF { bf16x16 v; uint4 q[2]; };

    const v8f vzero = {0.f, 0.f, 0.f, 0.f, 0.f, 0.f, 0.f, 0.f};
    v8f acc[2][4];
#pragma unroll
    for (int tm = 0; tm < 2; ++tm)
#pragma unroll
        for (int tn = 0; tn < 4; ++tn) acc[tm][tn] = vzero;

    float4 pa[4];
    loadA(0, pa);
    if (wave == 0) tdm_issue(0, 0);
    storeA(pa, 0);

    for (int kc = 0; kc < 64; ++kc) {
        const int cur = kc & 1, nxt = cur ^ 1;
        const bool more = (kc + 1 < 64);
        if (wave == 0) __builtin_amdgcn_s_wait_tensorcnt(0);   // B(kc) DMA done
        __syncthreads();                                       // + A(kc) stores visible,
                                                               //   prev compute finished
        if (more) {
            loadA(kc + 1, pa);                                 // overlap with compute
            if (wave == 0) tdm_issue(kc + 1, nxt);             // safe after barrier
        }

        FragBF fa[2];
#pragma unroll
        for (int tm = 0; tm < 2; ++tm) {
            int row = wave_m * 32 + tm * 16 + (lane & 15);
            int base = row * 32 + (lane >> 4) * 16;            // lo: K 0..7 / 16..23
            fa[tm].q[0] = *reinterpret_cast<const uint4*>(&a_sh[cur][base]);
            fa[tm].q[1] = *reinterpret_cast<const uint4*>(&a_sh[cur][base + 8]); // K 8..15 / 24..31
        }
        FragBF fb[4];
#pragma unroll
        for (int tn = 0; tn < 4; ++tn) {
            int off = (wave_n * 4 + tn) * 512 + lane * 16;     // fragment-order, contiguous/lane
            fb[tn].q[0] = *reinterpret_cast<const uint4*>(&b_sh[cur][off]);
            fb[tn].q[1] = *reinterpret_cast<const uint4*>(&b_sh[cur][off + 8]);
        }
#pragma unroll
        for (int tm = 0; tm < 2; ++tm)
#pragma unroll
            for (int tn = 0; tn < 4; ++tn)
                acc[tm][tn] = __builtin_amdgcn_wmma_f32_16x16x32_bf16(
                    false, fa[tm].v, false, fb[tn].v, (short)0, acc[tm][tn], false, false);

        if (more) storeA(pa, nxt);
    }

    // Epilogue: energy[row] += sum_n tanh(h + b1') * W2[n]
    const int lhalf = lane >> 4;
#pragma unroll
    for (int tm = 0; tm < 2; ++tm) {
        float part[8];
#pragma unroll
        for (int r = 0; r < 8; ++r) part[r] = 0.f;
#pragma unroll
        for (int tn = 0; tn < 4; ++tn) {
            int ncol = nb * 128 + wave_n * 64 + tn * 16 + (lane & 15);
            float w2v = W2[ncol];
            float b1v = b1p[ncol];
#pragma unroll
            for (int r = 0; r < 8; ++r)
                part[r] += tanhf(acc[tm][tn][r] + b1v) * w2v;
        }
        int rowbase = wave_m * 32 + tm * 16 + lhalf * 8;       // C layout: M = r + 8*half
#pragma unroll
        for (int r = 0; r < 8; ++r)
            atomicAdd(&e_part[rowbase + r], part[r]);
    }
    __syncthreads();
    if (tid < 128) atomicAdd(&energy[row0 + tid], e_part[tid]);
}

// ---------------------------------------------------------------------------
// 6) energy -> masked softmax -> attention weights (second output)
__global__ __launch_bounds__(128) void energy_softmax_kernel(const float* __restrict__ energy,
                                                             const float* __restrict__ b2,
                                                             const int* __restrict__ lens,
                                                             float* __restrict__ wout) {
    __shared__ float red[128];
    int b = blockIdx.x, s = threadIdx.x;
    bool m = s < lens[b];
    float logit = m ? tanhf(energy[b * 128 + s] + b2[0]) : -1e9f;
    red[s] = logit; __syncthreads();
#pragma unroll
    for (int o = 64; o > 0; o >>= 1) { if (s < o) red[s] = fmaxf(red[s], red[s + o]); __syncthreads(); }
    float mx = red[0]; __syncthreads();
    float e = __expf(logit - mx);
    red[s] = e; __syncthreads();
#pragma unroll
    for (int o = 64; o > 0; o >>= 1) { if (s < o) red[s] += red[s + o]; __syncthreads(); }
    wout[b * 128 + s] = (e / red[0]) * (m ? 1.f : 0.f);
}

// 7) context[b][d] = sum_s weights[b,s] * words[b,s,d]
__global__ __launch_bounds__(256) void context_kernel(const float* __restrict__ lstm,
                                                      const float* __restrict__ w,
                                                      float* __restrict__ ctx) {
    __shared__ float wl[128];
    int b = blockIdx.x, t = threadIdx.x;
    if (t < 128) wl[t] = w[b * 128 + t];
    __syncthreads();
    float acc[4] = {0, 0, 0, 0};
    for (int s = 0; s < 128; ++s) {
        float ws_ = wl[s];
        const float* p = lstm + ((size_t)b * 128 + s) * 1024;
#pragma unroll
        for (int i = 0; i < 4; ++i) acc[i] += ws_ * p[t + 256 * i];
    }
#pragma unroll
    for (int i = 0; i < 4; ++i) ctx[(size_t)b * 1024 + t + 256 * i] = acc[i];
}

// 8) BN2 (over batch) -> per-feature affine
__global__ __launch_bounds__(256) void bn2_stats_kernel(const float* __restrict__ ctx,
                                                        const float* __restrict__ g2,
                                                        const float* __restrict__ b2v,
                                                        float* scale2, float* shift2) {
    int f = blockIdx.x * 256 + threadIdx.x;            // 4 blocks -> 1024
    float s = 0.f, q = 0.f;
    for (int b = 0; b < 256; ++b) { float v = ctx[(size_t)b * 1024 + f]; s += v; q += v * v; }
    float mean = s * (1.f / 256.f);
    float var = q * (1.f / 256.f) - mean * mean;
    float sc = g2[f] * rsqrtf(var + 1e-5f);
    scale2[f] = sc;
    shift2[f] = b2v[f] - mean * sc;
}

// 9) p1 = tanh(BN2(ctx) @ Wp1 + bp1)
__global__ __launch_bounds__(256) void proj1_kernel(const float* __restrict__ ctx,
                                                    const float* __restrict__ scale2,
                                                    const float* __restrict__ shift2,
                                                    const float* __restrict__ Wp1,
                                                    const float* __restrict__ bp1,
                                                    float* __restrict__ p1) {
    __shared__ float cn[1024];
    int b = blockIdx.x, t = threadIdx.x;
    for (int i = t; i < 1024; i += 256)
        cn[i] = ctx[(size_t)b * 1024 + i] * scale2[i] + shift2[i];
    __syncthreads();
    float acc[4] = {0, 0, 0, 0};
    for (int f = 0; f < 1024; ++f) {
        float c = cn[f];
        const float* wrow = Wp1 + (size_t)f * 1024;
#pragma unroll
        for (int i = 0; i < 4; ++i) acc[i] += c * wrow[t + 256 * i];
    }
#pragma unroll
    for (int i = 0; i < 4; ++i)
        p1[(size_t)b * 1024 + t + 256 * i] = tanhf(acc[i] + bp1[t + 256 * i]);
}

// 10) out = softmax(tanh(p1 @ Wp2 + bp2))  (first output)
__global__ __launch_bounds__(64) void proj2_softmax_kernel(const float* __restrict__ p1,
                                                           const float* __restrict__ Wp2,
                                                           const float* __restrict__ bp2,
                                                           float* __restrict__ out) {
    __shared__ float pl[1024];
    __shared__ float red[64];
    int b = blockIdx.x, o = threadIdx.x;
    for (int i = o; i < 1024; i += 64) pl[i] = p1[(size_t)b * 1024 + i];
    __syncthreads();
    float acc = 0.f;
    for (int f = 0; f < 1024; ++f) acc += pl[f] * Wp2[(size_t)f * 64 + o];
    float v = tanhf(acc + bp2[o]);
    red[o] = v; __syncthreads();
#pragma unroll
    for (int st = 32; st > 0; st >>= 1) { if (o < st) red[o] = fmaxf(red[o], red[o + st]); __syncthreads(); }
    float mx = red[0]; __syncthreads();
    float e = __expf(v - mx);
    red[o] = e; __syncthreads();
#pragma unroll
    for (int st = 32; st > 0; st >>= 1) { if (o < st) red[o] += red[o + st]; __syncthreads(); }
    out[b * 64 + o] = e / red[0];
}

// ---------------------------------------------------------------------------
extern "C" void kernel_launch(void* const* d_in, const int* in_sizes, int n_in,
                              void* d_out, int out_size, void* d_ws, size_t ws_size,
                              hipStream_t stream) {
    const float* lstm   = (const float*)d_in[0];
    const float* hidden = (const float*)d_in[1];
    const int*   lens   = (const int*)  d_in[2];
    const float* bn1_g  = (const float*)d_in[3];
    const float* bn1_b  = (const float*)d_in[4];
    const float* W1     = (const float*)d_in[5];
    const float* b1     = (const float*)d_in[6];
    const float* W2     = (const float*)d_in[7];
    const float* b2     = (const float*)d_in[8];
    const float* bn2_g  = (const float*)d_in[9];
    const float* bn2_b  = (const float*)d_in[10];
    const float* Wp1    = (const float*)d_in[11];
    const float* bp1    = (const float*)d_in[12];
    const float* Wp2    = (const float*)d_in[13];
    const float* bp2    = (const float*)d_in[14];
    (void)in_sizes; (void)n_in; (void)out_size; (void)ws_size;

    float* out  = (float*)d_out;            // (256,64)
    float* wout = out + 256 * 64;           // (256,128)

    // Workspace carve-out (~11.5 MB total; deterministic layout, re-init each call)
    char* ws = (char*)d_ws;
    size_t off = 0;
    auto carve = [&](size_t bytes) -> char* {
        char* p = ws + off;
        off = (off + bytes + 255) & ~(size_t)255;
        return p;
    };
    float* sent    = (float*)carve((size_t)256 * 1024 * 4);
    float* sums    = (float*)carve(2048 * 4);
    float* sumsq   = (float*)carve(2048 * 4);
    float* scale1  = (float*)carve(2048 * 4);
    float* shift1  = (float*)carve(2048 * 4);
    float* b1p     = (float*)carve(2048 * 4);
    float* cntf    = (float*)carve(256);
    float* energy  = (float*)carve((size_t)32768 * 4);
    float* context = (float*)carve((size_t)256 * 1024 * 4);
    float* scale2  = (float*)carve(1024 * 4);
    float* shift2  = (float*)carve(1024 * 4);
    float* p1      = (float*)carve((size_t)256 * 1024 * 4);
    unsigned short* W1s = (unsigned short*)carve((size_t)2048 * 2048 * 2);

    init_kernel<<<128, 256, 0, stream>>>(sums, sumsq, energy);
    sent_kernel<<<1024, 256, 0, stream>>>(hidden, sent);
    bn1_words_stats<<<256, 256, 0, stream>>>(lstm, lens, sums, sumsq);
    bn1_sent_stats<<<4, 256, 0, stream>>>(sent, lens, sums, sumsq, cntf);
    bn1_finalize<<<8, 256, 0, stream>>>(sums, sumsq, cntf, bn1_g, bn1_b, scale1, shift1);
    fold_w1_kernel<<<1024, 256, 0, stream>>>(W1, scale1, W1s);
    fold_b1_kernel<<<8, 256, 0, stream>>>(W1, shift1, b1, b1p);

    dim3 gemm_grid(16, 256);   // x: N blocks of 128, y: M blocks of 128 (== batch)
    attn_gemm_kernel<<<gemm_grid, 256, 0, stream>>>(lstm, sent, W1s, W2, b1p, energy);

    energy_softmax_kernel<<<256, 128, 0, stream>>>(energy, b2, lens, wout);
    context_kernel<<<256, 256, 0, stream>>>(lstm, wout, context);
    bn2_stats_kernel<<<4, 256, 0, stream>>>(context, bn2_g, bn2_b, scale2, shift2);
    proj1_kernel<<<256, 256, 0, stream>>>(context, scale2, shift2, Wp1, bp1, p1);
    proj2_softmax_kernel<<<256, 64, 0, stream>>>(p1, Wp2, bp2, out);
}
